// DNCAE_36455682409004
// MI455X (gfx1250) — matrix operand order repaired and checked by reading, not compiled, not run on hardware
//
#include <hip/hip_runtime.h>
#include <hip/hip_bf16.h>
#include <math.h>

typedef __attribute__((ext_vector_type(16))) _Float16 v16h;
typedef __attribute__((ext_vector_type(8)))  float    v8f;

#define B_   64
#define T_   512
#define D_   512
#define V_   256
#define R_   4
#define W_   16
#define M_   5
#define RW_  64
#define IFC_ 135
#define XK_  576   // D + RW

// ---------- small device helpers ----------
__device__ inline float sigf(float x)      { return 1.f / (1.f + expf(-x)); }
__device__ inline float softplusf(float x) { return x > 20.f ? x : log1pf(expf(x)); }

__device__ inline v8f zf() {
  v8f a;
#pragma unroll
  for (int i = 0; i < 8; ++i) a[i] = 0.f;
  return a;
}

// A fragment: 16xK tile from row-major (M x K) f16 matrix.
// A layout (contiguous k order, matches ldB): lane<16 holds row=lane, K {k0..k0+15};
// lane>=16 holds row=lane-16, K {k0+16..k0+31}.
__device__ inline v16h ldA(const _Float16* p0, int ld, int row0, int k0, int lane) {
  const _Float16* p = p0 + (size_t)(row0 + (lane & 15)) * ld + k0 + ((lane >> 4) << 4);
  v16h r;
#pragma unroll
  for (int i = 0; i < 16; ++i) r[i] = p[i];
  return r;
}

// B fragment from row-major (N x K) matrix (activations, batch-major):
// lane&15 = column (batch), 16 contiguous K values starting at k0 (+16 for lanes>=16).
__device__ inline v16h ldB(const _Float16* p0, int ld, int col0, int k0, int lane) {
  const _Float16* p = p0 + (size_t)(col0 + (lane & 15)) * ld + k0 + ((lane >> 4) << 4);
  v16h r;
#pragma unroll
  for (int i = 0; i < 16; ++i) r[i] = p[i];
  return r;
}

__device__ inline v8f wmma16(v16h a, v16h b, v8f c) {
  return __builtin_amdgcn_wmma_f32_16x16x32_f16(false, a, false, b, (short)0, c, false, false);
}

// ---------- one-time prep kernels ----------
__global__ void k_zero(uint32_t* p, int n) {
  int i = blockIdx.x * 256 + threadIdx.x;
  if (i < n) p[i] = 0u;
}

__global__ void k_cvt(const float* s, _Float16* d, int n) {
  int i = blockIdx.x * 256 + threadIdx.x;
  if (i < n) d[i] = (_Float16)s[i];
}

// W_if (135x512) -> zero-padded (144x512) f16
__global__ void k_wif(const float* s, _Float16* d) {
  int i = blockIdx.x * 256 + threadIdx.x;
  if (i >= 144 * 512) return;
  int r = i >> 9, c = i & 511;
  d[i] = (r < IFC_) ? (_Float16)s[r * 512 + c] : (_Float16)0.f;
}

// ---------- per-step kernels ----------
__global__ void k_embed(const int* tok, const float* emb, _Float16* xin, int t) {
  int b = blockIdx.x;
  int tk = tok[b * T_ + t];
  const float* e = emb + (size_t)tk * D_;
  for (int d = threadIdx.x; d < D_; d += blockDim.x)
    xin[b * XK_ + d] = (_Float16)e[d];
}

// LSTM layer 0: block = 4 waves (one per gate i,f,g,o), one 16-unit x 16-batch tile.
__global__ void k_lstm0(const _Float16* Wi, const _Float16* Wh, const float* bias,
                        const _Float16* xin, const _Float16* hin,
                        float* cst, _Float16* hout) {
  __shared__ float gl[4][16][17];
  int lane = threadIdx.x & 31, wv = threadIdx.x >> 5;
  int ut = blockIdx.x, bt = blockIdx.y;
  int row0 = wv * D_ + ut * 16;
  v8f acc = zf();
  for (int k0 = 0; k0 < XK_; k0 += 32) {
    __builtin_prefetch(Wi + (size_t)row0 * XK_ + k0 + 64, 0, 1);
    v16h a = ldA(Wi, XK_, row0, k0, lane);
    v16h b = ldB(xin, XK_, bt * 16, k0, lane);
    acc = wmma16(a, b, acc);
  }
  for (int k0 = 0; k0 < D_; k0 += 32) {
    __builtin_prefetch(Wh + (size_t)row0 * D_ + k0 + 64, 0, 1);
    v16h a = ldA(Wh, D_, row0, k0, lane);
    v16h b = ldB(hin, D_, bt * 16, k0, lane);
    acc = wmma16(a, b, acc);
  }
  int n = lane & 15, mb = (lane >> 4) * 8;
#pragma unroll
  for (int r = 0; r < 8; ++r) gl[wv][mb + r][n] = acc[r];
  __syncthreads();
  for (int e = threadIdx.x; e < 256; e += blockDim.x) {
    int m = e >> 4, nn = e & 15;
    int unit = ut * 16 + m, b = bt * 16 + nn;
    float gi = gl[0][m][nn] + bias[unit];
    float gf = gl[1][m][nn] + bias[D_ + unit];
    float gg = gl[2][m][nn] + bias[2 * D_ + unit];
    float go = gl[3][m][nn] + bias[3 * D_ + unit];
    float c = sigf(gf) * cst[b * D_ + unit] + sigf(gi) * tanhf(gg);
    float h = sigf(go) * tanhf(c);
    cst[b * D_ + unit] = c;
    hout[b * D_ + unit] = (_Float16)h;
  }
}

// LSTM layer 1: also writes clipped output into yin[:, :512]
__global__ void k_lstm1(const _Float16* Wi, const _Float16* Wh, const float* bias,
                        const _Float16* x0, const _Float16* hin,
                        float* cst, _Float16* hout, _Float16* yin) {
  __shared__ float gl[4][16][17];
  int lane = threadIdx.x & 31, wv = threadIdx.x >> 5;
  int ut = blockIdx.x, bt = blockIdx.y;
  int row0 = wv * D_ + ut * 16;
  v8f acc = zf();
  for (int k0 = 0; k0 < D_; k0 += 32) {
    __builtin_prefetch(Wi + (size_t)row0 * D_ + k0 + 64, 0, 1);
    v16h a = ldA(Wi, D_, row0, k0, lane);
    v16h b = ldB(x0, D_, bt * 16, k0, lane);
    acc = wmma16(a, b, acc);
  }
  for (int k0 = 0; k0 < D_; k0 += 32) {
    __builtin_prefetch(Wh + (size_t)row0 * D_ + k0 + 64, 0, 1);
    v16h a = ldA(Wh, D_, row0, k0, lane);
    v16h b = ldB(hin, D_, bt * 16, k0, lane);
    acc = wmma16(a, b, acc);
  }
  int n = lane & 15, mb = (lane >> 4) * 8;
#pragma unroll
  for (int r = 0; r < 8; ++r) gl[wv][mb + r][n] = acc[r];
  __syncthreads();
  for (int e = threadIdx.x; e < 256; e += blockDim.x) {
    int m = e >> 4, nn = e & 15;
    int unit = ut * 16 + m, b = bt * 16 + nn;
    float gi = gl[0][m][nn] + bias[unit];
    float gf = gl[1][m][nn] + bias[D_ + unit];
    float gg = gl[2][m][nn] + bias[2 * D_ + unit];
    float go = gl[3][m][nn] + bias[3 * D_ + unit];
    float c = sigf(gf) * cst[b * D_ + unit] + sigf(gi) * tanhf(gg);
    float h = sigf(go) * tanhf(c);
    cst[b * D_ + unit] = c;
    hout[b * D_ + unit] = (_Float16)h;               // carry keeps unclipped nh1
    float o = fminf(fmaxf(h, -20.f), 20.f);
    yin[b * XK_ + unit] = (_Float16)o;               // clipped out for xi / y
  }
}

// Interface GEMM (135x64, K=512) + scalar DNC memory update (1 thread / batch elem).
__global__ void k_ifdnc(const _Float16* Wif, const float* bif,
                        _Float16* yin, _Float16* xin,
                        float* mem, float* link, float* prec,
                        float* rwp, float* wwp, float* usg) {
  __shared__ float xs[144][16];
  int lane = threadIdx.x & 31, wv = threadIdx.x >> 5;
  int bt = blockIdx.x;
  for (int rt = wv; rt < 9; rt += 8) {
    v8f acc = zf();
    int row0 = rt * 16;
    for (int k0 = 0; k0 < D_; k0 += 32) {
      v16h a = ldA(Wif, D_, row0, k0, lane);
      v16h b = ldB(yin, XK_, bt * 16, k0, lane);
      acc = wmma16(a, b, acc);
    }
    int n = lane & 15, mb = (lane >> 4) * 8;
#pragma unroll
    for (int r = 0; r < 8; ++r) {
      int row = row0 + mb + r;
      xs[row][n] = acc[r] + (row < IFC_ ? bif[row] : 0.f);
    }
  }
  __syncthreads();

  if (threadIdx.x < 16) {
    int b = bt * 16 + threadIdx.x;
    float xi[IFC_];
    for (int k = 0; k < IFC_; ++k) xi[k] = xs[k][threadIdx.x];
    float* Mm = mem  + b * (M_ * W_);
    float* Lk = link + b * (M_ * M_);
    float* Pc = prec + b * M_;
    float* Rp = rwp  + b * (R_ * M_);
    float* Wp = wwp  + b * M_;
    float* Ug = usg  + b * M_;

    float rk[R_][W_], rs[R_], wk[W_], er[W_], wvv[W_], fg[R_], md[R_][3];
    for (int r = 0; r < R_; ++r)
      for (int k = 0; k < W_; ++k) rk[r][k] = tanhf(xi[r * W_ + k]);
    for (int r = 0; r < R_; ++r) rs[r] = softplusf(xi[64 + r]);
    for (int k = 0; k < W_; ++k) wk[k] = tanhf(xi[68 + k]);
    float wstr = softplusf(xi[84]);
    for (int k = 0; k < W_; ++k) er[k]  = sigf(xi[85 + k]);
    for (int k = 0; k < W_; ++k) wvv[k] = tanhf(xi[101 + k]);
    for (int r = 0; r < R_; ++r) fg[r]  = sigf(xi[117 + r]);
    float ag = sigf(xi[121]), wg = sigf(xi[122]);
    for (int r = 0; r < R_; ++r) {
      float m0 = xi[123 + r * 3], m1 = xi[124 + r * 3], m2 = xi[125 + r * 3];
      float mx = fmaxf(m0, fmaxf(m1, m2));
      float e0 = expf(m0 - mx), e1 = expf(m1 - mx), e2 = expf(m2 - mx);
      float s = e0 + e1 + e2;
      md[r][0] = e0 / s; md[r][1] = e1 / s; md[r][2] = e2 / s;
    }
    // usage update
    float u2[M_];
    for (int m = 0; m < M_; ++m) {
      float u = Ug[m] + (1.f - Ug[m]) * Wp[m];
      float psi = 1.f;
      for (int r = 0; r < R_; ++r) psi *= (1.f - fg[r] * Rp[r * M_ + m]);
      u2[m] = u * psi;
    }
    // write content weighting
    float nk = 0.f;
    for (int k = 0; k < W_; ++k) nk += wk[k] * wk[k];
    nk = sqrtf(nk) + 1e-6f;
    float sc[M_], mxs = -1e30f;
    for (int m = 0; m < M_; ++m) {
      float d = 0.f, nm = 0.f;
      for (int k = 0; k < W_; ++k) { float v = Mm[m * W_ + k]; d += v * wk[k]; nm += v * v; }
      sc[m] = d / ((sqrtf(nm) + 1e-6f) * nk) * wstr;
      mxs = fmaxf(mxs, sc[m]);
    }
    float ssum = 0.f, wcw[M_];
    for (int m = 0; m < M_; ++m) { wcw[m] = expf(sc[m] - mxs); ssum += wcw[m]; }
    for (int m = 0; m < M_; ++m) wcw[m] /= ssum;
    // allocation (sort + exclusive cumprod + unsort)
    float su[M_]; int id[M_];
    for (int m = 0; m < M_; ++m) { su[m] = 1e-6f + (1.f - 1e-6f) * u2[m]; id[m] = m; }
    for (int i = 1; i < M_; ++i) {
      float v = su[i]; int j = id[i]; int p = i - 1;
      while (p >= 0 && su[p] > v) { su[p + 1] = su[p]; id[p + 1] = id[p]; --p; }
      su[p + 1] = v; id[p + 1] = j;
    }
    float al[M_]; float prod = 1.f;
    for (int j = 0; j < M_; ++j) { al[id[j]] = (1.f - su[j]) * prod; prod *= su[j]; }
    // write weights + memory/link/precedence update
    float wwv[M_], wsum = 0.f;
    for (int m = 0; m < M_; ++m) { wwv[m] = wg * (ag * al[m] + (1.f - ag) * wcw[m]); wsum += wwv[m]; }
    float m2a[M_][W_];
    for (int m = 0; m < M_; ++m)
      for (int k = 0; k < W_; ++k)
        m2a[m][k] = Mm[m * W_ + k] * (1.f - wwv[m] * er[k]) + wwv[m] * wvv[k];
    float L2[M_][M_];
    for (int i = 0; i < M_; ++i)
      for (int j = 0; j < M_; ++j)
        L2[i][j] = (i == j) ? 0.f
                            : ((1.f - wwv[i] - wwv[j]) * Lk[i * M_ + j] + wwv[i] * Pc[j]);
    float P2[M_];
    for (int m = 0; m < M_; ++m) P2[m] = (1.f - wsum) * Pc[m] + wwv[m];
    // read heads
    float mnorm[M_];
    for (int m = 0; m < M_; ++m) {
      float s = 0.f;
      for (int k = 0; k < W_; ++k) s += m2a[m][k] * m2a[m][k];
      mnorm[m] = sqrtf(s) + 1e-6f;
    }
    float nrw[R_][M_];
    for (int r = 0; r < R_; ++r) {
      float kn2 = 0.f;
      for (int k = 0; k < W_; ++k) kn2 += rk[r][k] * rk[r][k];
      kn2 = sqrtf(kn2) + 1e-6f;
      float s2[M_], mx2 = -1e30f;
      for (int m = 0; m < M_; ++m) {
        float d = 0.f;
        for (int k = 0; k < W_; ++k) d += m2a[m][k] * rk[r][k];
        s2[m] = d / (mnorm[m] * kn2) * rs[r];
        mx2 = fmaxf(mx2, s2[m]);
      }
      float sm = 0.f, rcw[M_];
      for (int m = 0; m < M_; ++m) { rcw[m] = expf(s2[m] - mx2); sm += rcw[m]; }
      for (int m = 0; m < M_; ++m) rcw[m] /= sm;
      float fwdv[M_], bwdv[M_];
      for (int i = 0; i < M_; ++i) {
        float s = 0.f;
        for (int j = 0; j < M_; ++j) s += L2[i][j] * Rp[r * M_ + j];
        fwdv[i] = s;
      }
      for (int j = 0; j < M_; ++j) {
        float s = 0.f;
        for (int i = 0; i < M_; ++i) s += L2[i][j] * Rp[r * M_ + i];
        bwdv[j] = s;
      }
      for (int m = 0; m < M_; ++m)
        nrw[r][m] = md[r][0] * bwdv[m] + md[r][1] * fwdv[m] + md[r][2] * rcw[m];
    }
    // read vectors -> yin tail (for y) and xin tail (last_read for next step)
    for (int r = 0; r < R_; ++r)
      for (int k = 0; k < W_; ++k) {
        float s = 0.f;
        for (int m = 0; m < M_; ++m) s += nrw[r][m] * m2a[m][k];
        _Float16 hv = (_Float16)s;
        yin[b * XK_ + D_ + r * W_ + k] = hv;
        xin[b * XK_ + D_ + r * W_ + k] = hv;
      }
    // commit state
    for (int m = 0; m < M_; ++m)
      for (int k = 0; k < W_; ++k) Mm[m * W_ + k] = m2a[m][k];
    for (int i = 0; i < M_; ++i)
      for (int j = 0; j < M_; ++j) Lk[i * M_ + j] = L2[i][j];
    for (int m = 0; m < M_; ++m) { Pc[m] = P2[m]; Wp[m] = wwv[m]; Ug[m] = u2[m]; }
    for (int r = 0; r < R_; ++r)
      for (int m = 0; m < M_; ++m) Rp[r * M_ + m] = nrw[r][m];
  }
}

// y = [out, read_vecs] @ W_out^T + b_out, stored f16 into Y (B,T,D)
__global__ void k_yout(const _Float16* Wo, const float* bo, const _Float16* yin,
                       _Float16* Y, int t) {
  int lane = threadIdx.x & 31, wv = threadIdx.x >> 5;
  int rt = blockIdx.x * 4 + wv, bt = blockIdx.y;
  int row0 = rt * 16;
  v8f acc = zf();
  for (int k0 = 0; k0 < XK_; k0 += 32) {
    __builtin_prefetch(Wo + (size_t)row0 * XK_ + k0 + 64, 0, 1);
    v16h a = ldA(Wo, XK_, row0, k0, lane);
    v16h b = ldB(yin, XK_, bt * 16, k0, lane);
    acc = wmma16(a, b, acc);
  }
  int n = lane & 15, mb = (lane >> 4) * 8;
#pragma unroll
  for (int r = 0; r < 8; ++r) {
    int d = row0 + mb + r;
    int b = bt * 16 + n;
    Y[((size_t)(b * T_ + t)) * D_ + d] = (_Float16)(acc[r] + bo[d]);
  }
}

// logits = Y @ W_fc^T + b_fc, transposed store to (B, V, T)
__global__ void k_logits(const _Float16* Wfc, const float* bfc, const _Float16* Y,
                         float* out) {
  int lane = threadIdx.x & 31;
  int vt = blockIdx.x, rt = blockIdx.y;
  int v0 = vt * 16, r0 = rt * 16;
  v8f acc = zf();
  for (int k0 = 0; k0 < D_; k0 += 32) {
    v16h a = ldA(Wfc, D_, v0, k0, lane);
    v16h b = ldB(Y, D_, r0, k0, lane);
    acc = wmma16(a, b, acc);
  }
  int n = lane & 15, mb = (lane >> 4) * 8;
#pragma unroll
  for (int r = 0; r < 8; ++r) {
    int v = v0 + mb + r;
    int row = r0 + n;                 // row = b*T + t
    int b = row >> 9, t = row & 511;  // T_ == 512
    out[(size_t)b * V_ * T_ + (size_t)v * T_ + t] = acc[r] + bfc[v];
  }
}

// ---------- host orchestration ----------
extern "C" void kernel_launch(void* const* d_in, const int* in_sizes, int n_in,
                              void* d_out, int out_size, void* d_ws, size_t ws_size,
                              hipStream_t stream) {
  (void)in_sizes; (void)n_in; (void)out_size; (void)ws_size;
  const int*   tok  = (const int*)d_in[0];
  const float* emb  = (const float*)d_in[1];
  const float* Wi0  = (const float*)d_in[2];
  const float* Wh0  = (const float*)d_in[3];
  const float* b0   = (const float*)d_in[4];
  const float* Wi1  = (const float*)d_in[5];
  const float* Wh1  = (const float*)d_in[6];
  const float* b1   = (const float*)d_in[7];
  const float* Wif  = (const float*)d_in[8];
  const float* bif  = (const float*)d_in[9];
  const float* Wout = (const float*)d_in[10];
  const float* bout = (const float*)d_in[11];
  const float* Wfc  = (const float*)d_in[12];
  const float* bfc  = (const float*)d_in[13];
  float* out = (float*)d_out;
  char* w = (char*)d_ws;

  size_t off = 0;
  auto alc = [&](size_t n) { size_t o = off; off = (off + n + 255) & ~(size_t)255; return o; };

  // --- state block (zeroed each call; lives at ws start) ---
  size_t o_c0  = alc((size_t)B_ * D_ * 4);
  size_t o_c1  = alc((size_t)B_ * D_ * 4);
  size_t o_mem = alc((size_t)B_ * M_ * W_ * 4);
  size_t o_lnk = alc((size_t)B_ * M_ * M_ * 4);
  size_t o_prc = alc((size_t)B_ * M_ * 4);
  size_t o_rwp = alc((size_t)B_ * R_ * M_ * 4);
  size_t o_wwp = alc((size_t)B_ * M_ * 4);
  size_t o_usg = alc((size_t)B_ * M_ * 4);
  size_t o_h0a = alc((size_t)B_ * D_ * 2);
  size_t o_h0b = alc((size_t)B_ * D_ * 2);
  size_t o_h1a = alc((size_t)B_ * D_ * 2);
  size_t o_h1b = alc((size_t)B_ * D_ * 2);
  size_t o_xin = alc((size_t)B_ * XK_ * 2);
  size_t o_yin = alc((size_t)B_ * XK_ * 2);
  size_t state_end = off;                        // bytes to zero (256-aligned)

  // --- f16 weights + Y buffer ---
  size_t o_Wi0 = alc((size_t)2048 * XK_ * 2);
  size_t o_Wh0 = alc((size_t)2048 * D_ * 2);
  size_t o_Wi1 = alc((size_t)2048 * D_ * 2);
  size_t o_Wh1 = alc((size_t)2048 * D_ * 2);
  size_t o_Wif = alc((size_t)144 * D_ * 2);
  size_t o_Wou = alc((size_t)D_ * XK_ * 2);
  size_t o_Wfc = alc((size_t)V_ * D_ * 2);
  size_t o_Y   = alc((size_t)B_ * T_ * D_ * 2);

  float*     pC0  = (float*)(w + o_c0);
  float*     pC1  = (float*)(w + o_c1);
  float*     pMem = (float*)(w + o_mem);
  float*     pLnk = (float*)(w + o_lnk);
  float*     pPrc = (float*)(w + o_prc);
  float*     pRwp = (float*)(w + o_rwp);
  float*     pWwp = (float*)(w + o_wwp);
  float*     pUsg = (float*)(w + o_usg);
  _Float16*  pH0a = (_Float16*)(w + o_h0a);
  _Float16*  pH0b = (_Float16*)(w + o_h0b);
  _Float16*  pH1a = (_Float16*)(w + o_h1a);
  _Float16*  pH1b = (_Float16*)(w + o_h1b);
  _Float16*  pXin = (_Float16*)(w + o_xin);
  _Float16*  pYin = (_Float16*)(w + o_yin);
  _Float16*  hWi0 = (_Float16*)(w + o_Wi0);
  _Float16*  hWh0 = (_Float16*)(w + o_Wh0);
  _Float16*  hWi1 = (_Float16*)(w + o_Wi1);
  _Float16*  hWh1 = (_Float16*)(w + o_Wh1);
  _Float16*  hWif = (_Float16*)(w + o_Wif);
  _Float16*  hWou = (_Float16*)(w + o_Wou);
  _Float16*  hWfc = (_Float16*)(w + o_Wfc);
  _Float16*  pY   = (_Float16*)(w + o_Y);

  // zero recurrent/DNC state (deterministic across calls)
  int zw = (int)(state_end / 4);
  k_zero<<<(zw + 255) / 256, 256, 0, stream>>>((uint32_t*)w, zw);

  // convert weights to f16 (stays resident in L2 for the whole scan)
  struct { const float* s; _Float16* d; int n; } cv[] = {
    { Wi0,  hWi0, 2048 * XK_ }, { Wh0, hWh0, 2048 * D_ },
    { Wi1,  hWi1, 2048 * D_  }, { Wh1, hWh1, 2048 * D_ },
    { Wout, hWou, D_ * XK_   }, { Wfc, hWfc, V_ * D_   },
  };
  for (auto& c : cv)
    k_cvt<<<(c.n + 255) / 256, 256, 0, stream>>>(c.s, c.d, c.n);
  k_wif<<<(144 * 512 + 255) / 256, 256, 0, stream>>>(Wif, hWif);

  for (int t = 0; t < T_; ++t) {
    _Float16* h0in  = (t & 1) ? pH0b : pH0a;
    _Float16* h0out = (t & 1) ? pH0a : pH0b;
    _Float16* h1in  = (t & 1) ? pH1b : pH1a;
    _Float16* h1out = (t & 1) ? pH1a : pH1b;
    k_embed<<<B_, 256, 0, stream>>>(tok, emb, pXin, t);
    k_lstm0<<<dim3(32, 4), 128, 0, stream>>>(hWi0, hWh0, b0, pXin, h0in, pC0, h0out);
    k_lstm1<<<dim3(32, 4), 128, 0, stream>>>(hWi1, hWh1, b1, h0out, h1in, pC1, h1out, pYin);
    k_ifdnc<<<4, 256, 0, stream>>>(hWif, bif, pYin, pXin,
                                   pMem, pLnk, pPrc, pRwp, pWwp, pUsg);
    k_yout<<<dim3(8, 4), 128, 0, stream>>>(hWou, bout, pYin, pY, t);
  }

  k_logits<<<dim3(V_ / 16, (B_ * T_) / 16), 32, 0, stream>>>(hWfc, bfc, pY, out);
}
